// SiameseCNN_9861244912182
// MI455X (gfx1250) — matrix operand order repaired and checked by reading, not compiled, not run on hardware
//
#include <hip/hip_runtime.h>

typedef _Float16 h16;
typedef __attribute__((ext_vector_type(16))) _Float16 v16h;
typedef __attribute__((ext_vector_type(8)))  _Float16 h8;
typedef __attribute__((ext_vector_type(8)))  float    v8f;

#define EPS 1e-5f

// ---- workspace byte offsets (256B aligned) ----
#define OFF_ACT1  0UL              // 1536*128*64 f16 (also reused as act3)
#define OFF_ACT2  25165824UL       // 1536*128*64 f16 (also reused as act4)
#define OFF_FCB1  50331648UL       // 1536*512 f16
#define OFF_FCB2  51904512UL       // 1536*512 f16
#define OFF_FCB3  53477376UL       // 1536*128 f16
#define OFF_C2W   53870592UL       // 9*64*64 f16   [tap][o][c]
#define OFF_LC1W  53944320UL       // 128*64*64 f16 [p][o][c]
#define OFF_LC2W  54992896UL       // 128*64*64 f16 [p][o][c]
#define OFF_FC1W  56041472UL       // 512*8192 f16  [n][k]
#define OFF_FC2W  64430080UL       // 512*512 f16
#define OFF_FC3W  64954368UL       // 128*512 f16
#define OFF_OUTW  65085440UL       // 128*128 f16
#define OFF_P     65118208UL       // folded bn scale/shift (f32)

// float indices inside the P region
#define P_S1 0
#define P_T1 64
#define P_S2 128
#define P_T2 192
#define P_S3 256
#define P_T3 320
#define P_S4 384
#define P_T4 448
#define P_SF1 512
#define P_TF1 1024
#define P_SF2 1536
#define P_TF2 2048
#define P_SF3 2560
#define P_TF3 2688
#define P_SO 2816
#define P_TO 2944
#define P_S0 3072
#define P_T0 3073

// ---------------- WMMA fragment helpers (per CDNA5 ISA layout tables) ----------------
// A 16x32 f16: lane<16 -> row=lane, e[0..7]=K[k0..k0+7],  e[8..15]=K[k0+16..k0+23]
//              lane>=16 -> row=lane-16, e[0..7]=K[k0+8..], e[8..15]=K[k0+24..]
__device__ __forceinline__ v16h load_a_frag(const h16* __restrict__ A, int lda, int k0, int lane) {
    int m  = lane & 15;
    int kh = (lane >> 4) << 3;   // 0 or 8
    const h16* p = A + (size_t)m * lda + k0 + kh;
    h8 lo = *(const h8*)(p);
    h8 hi = *(const h8*)(p + 16);
    v16h r;
#pragma unroll
    for (int i = 0; i < 8; ++i) { r[i] = lo[i]; r[i + 8] = hi[i]; }
    return r;
}

// B 32x16 f16 from Bt=[n][k]: lane<16 -> col=lane, e[0..7]=Bt[n][k0..k0+7], e[8..15]=Bt[n][k0+16..k0+23]
//                             lane>=16 -> col=lane-16, e[0..7]=Bt[n][k0+8..], e[8..15]=Bt[n][k0+24..]
__device__ __forceinline__ v16h load_b_frag(const h16* __restrict__ Bt, int ldb, int k0, int lane) {
    int n  = lane & 15;
    int kh = (lane >> 4) << 3;   // 0 or 8
    const h16* p = Bt + (size_t)n * ldb + k0 + kh;
    h8 lo = *(const h8*)(p);
    h8 hi = *(const h8*)(p + 16);
    v16h r;
#pragma unroll
    for (int i = 0; i < 8; ++i) { r[i] = lo[i]; r[i + 8] = hi[i]; }
    return r;
}

// ---------------- parameter folding: bn -> y = x*s + t (biases folded) ----------------
__global__ void prep_params(
    const float* bn0_g, const float* bn0_b, const float* bn0_m, const float* bn0_v,
    const float* conv1_b, const float* bn1_g, const float* bn1_b, const float* bn1_m, const float* bn1_v,
    const float* conv2_b, const float* bn2_g, const float* bn2_b, const float* bn2_m, const float* bn2_v,
    const float* bn3_g, const float* bn3_b, const float* bn3_m, const float* bn3_v,
    const float* bn4_g, const float* bn4_b, const float* bn4_m, const float* bn4_v,
    const float* fc1_b, const float* bnf1_g, const float* bnf1_b, const float* bnf1_m, const float* bnf1_v,
    const float* fc2_b, const float* bnf2_g, const float* bnf2_b, const float* bnf2_m, const float* bnf2_v,
    const float* fc3_b, const float* bnf3_g, const float* bnf3_b, const float* bnf3_m, const float* bnf3_v,
    const float* out_b, float* P)
{
    int tid = threadIdx.x;
    if (tid == 0) {
        float s = bn0_g[0] * rsqrtf(bn0_v[0] + EPS);
        P[P_S0] = s; P[P_T0] = bn0_b[0] - bn0_m[0] * s;
    }
    for (int c = tid; c < 64; c += blockDim.x) {
        float s;
        s = bn1_g[c] * rsqrtf(bn1_v[c] + EPS); P[P_S1 + c] = s; P[P_T1 + c] = conv1_b[c] * s + bn1_b[c] - bn1_m[c] * s;
        s = bn2_g[c] * rsqrtf(bn2_v[c] + EPS); P[P_S2 + c] = s; P[P_T2 + c] = conv2_b[c] * s + bn2_b[c] - bn2_m[c] * s;
        s = bn3_g[c] * rsqrtf(bn3_v[c] + EPS); P[P_S3 + c] = s; P[P_T3 + c] = bn3_b[c] - bn3_m[c] * s;
        s = bn4_g[c] * rsqrtf(bn4_v[c] + EPS); P[P_S4 + c] = s; P[P_T4 + c] = bn4_b[c] - bn4_m[c] * s;
    }
    for (int c = tid; c < 512; c += blockDim.x) {
        float s;
        s = bnf1_g[c] * rsqrtf(bnf1_v[c] + EPS); P[P_SF1 + c] = s; P[P_TF1 + c] = fc1_b[c] * s + bnf1_b[c] - bnf1_m[c] * s;
        s = bnf2_g[c] * rsqrtf(bnf2_v[c] + EPS); P[P_SF2 + c] = s; P[P_TF2 + c] = fc2_b[c] * s + bnf2_b[c] - bnf2_m[c] * s;
    }
    for (int c = tid; c < 128; c += blockDim.x) {
        float s = bnf3_g[c] * rsqrtf(bnf3_v[c] + EPS);
        P[P_SF3 + c] = s; P[P_TF3 + c] = fc3_b[c] * s + bnf3_b[c] - bnf3_m[c] * s;
        P[P_SO + c] = 1.0f; P[P_TO + c] = out_b[c];
    }
}

// ---------------- weight repacks (f32 -> f16) ----------------
__global__ void pack_conv2(const float* __restrict__ w, h16* __restrict__ out) {
    int i = blockIdx.x * 256 + threadIdx.x;        // [tap][o][c], 9*64*64
    if (i >= 9 * 64 * 64) return;
    int t = i >> 12, o = (i >> 6) & 63, c = i & 63;
    out[i] = (h16)w[(o * 64 + c) * 9 + t];         // src (o,c,ky,kx)
}
__global__ void pack_lc(const float* __restrict__ w, h16* __restrict__ out) {
    int i = blockIdx.x * 256 + threadIdx.x;        // [p][o][c], 128*64*64
    int p = i >> 12, o = (i >> 6) & 63, c = i & 63;
    out[i] = (h16)w[((o * 64 + c) << 7) + p];      // src (o,c,h,w), p=h*8+w
}
__global__ void cvt_f16(const float* __restrict__ in, h16* __restrict__ out, int n) {
    int i = blockIdx.x * 256 + threadIdx.x;
    if (i < n) out[i] = (h16)in[i];
}

// ---------------- stage 1: bn0 + conv1(1->64) + bn1 + relu, channels-last f16 out ----------------
__global__ __launch_bounds__(128) void conv1_kernel(
    const float* __restrict__ anchor, const float* __restrict__ positive,
    const float* __restrict__ negative, const float* __restrict__ w1,
    const float* __restrict__ P, h16* __restrict__ act1)
{
    __shared__ float xs[128];
    __shared__ float wsm[576];
    __shared__ float s1[64], t1[64];
    int img = blockIdx.x;                 // 0..1535
    int tid = threadIdx.x;                // 0..127 = position
    const float* x = (img < 512)  ? anchor   + (size_t)img * 128
                   : (img < 1024) ? positive + (size_t)(img - 512) * 128
                                  : negative + (size_t)(img - 1024) * 128;
    float s0 = P[P_S0], t0 = P[P_T0];
    xs[tid] = x[tid] * s0 + t0;           // bn0 before zero-padded conv
    for (int i = tid; i < 576; i += 128) wsm[i] = w1[i];
    if (tid < 64) { s1[tid] = P[P_S1 + tid]; t1[tid] = P[P_T1 + tid]; }
    __syncthreads();
    int h = tid >> 3, w = tid & 7;
    h16* outp = act1 + ((size_t)img * 128 + tid) * 64;
    for (int o = 0; o < 64; ++o) {
        float acc = 0.f;
#pragma unroll
        for (int t = 0; t < 9; ++t) {
            int hh = h + t / 3 - 1, ww = w + t % 3 - 1;
            if ((unsigned)hh < 16u && (unsigned)ww < 8u)
                acc += wsm[o * 9 + t] * xs[hh * 8 + ww];
        }
        float v = acc * s1[o] + t1[o];
        outp[o] = (h16)fmaxf(v, 0.f);
    }
}

// ---------------- stage 2: conv2 as implicit GEMM (9 taps x K=64) + bn2 + relu ----------------
// act1: [(b*128+p)][c]; out act2: [p][b][c]   (one wave = 16 positions x 64 out-ch)
__global__ __launch_bounds__(128) void conv2_wmma(
    const h16* __restrict__ act1, const h16* __restrict__ W,   // W: [tap][o][c]
    const float* __restrict__ S, const float* __restrict__ T,
    h16* __restrict__ act2)
{
    int lane = threadIdx.x & 31;
    int wid  = threadIdx.x >> 5;
    int mt   = blockIdx.x * 4 + wid;       // 0..12287
    int b    = mt >> 3;
    int p0   = (mt & 7) * 16;
    int m    = lane & 15;
    int p    = p0 + m;
    int h    = p >> 3, w = p & 7;
    int kh   = (lane >> 4) << 3;

    v8f acc[4] = {};
#pragma unroll
    for (int t = 0; t < 9; ++t) {
        int dy = t / 3 - 1, dx = t % 3 - 1;
        int hh = h + dy, ww = w + dx;
        bool valid = ((unsigned)hh < 16u) && ((unsigned)ww < 8u);
        const h16* arow = act1 + ((size_t)b * 128 + (size_t)(hh * 8 + ww)) * 64;
        const h16* Wt   = W + t * 4096;
#pragma unroll
        for (int k0 = 0; k0 < 64; k0 += 32) {
            v16h a = {};
            if (valid) {
                const h16* q = arow + k0 + kh;
                h8 lo = *(const h8*)q;
                h8 hi = *(const h8*)(q + 16);
#pragma unroll
                for (int i = 0; i < 8; ++i) { a[i] = lo[i]; a[i + 8] = hi[i]; }
            }
#pragma unroll
            for (int j = 0; j < 4; ++j) {
                v16h bf = load_b_frag(Wt + (size_t)(j * 16) * 64, 64, k0, lane);
                acc[j] = __builtin_amdgcn_wmma_f32_16x16x32_f16(
                    false, a, false, bf, (short)0, acc[j], false, false);
            }
        }
    }
    int ncol = lane & 15;
    int rhi  = (lane >> 4) * 8;
#pragma unroll
    for (int j = 0; j < 4; ++j) {
#pragma unroll
        for (int r = 0; r < 8; ++r) {
            int o  = j * 16 + ncol;
            int pr = p0 + rhi + r;
            float v = acc[j][r] * S[o] + T[o];
            v = fmaxf(v, 0.f);
            act2[((size_t)pr * 1536 + b) * 64 + o] = (h16)v;
        }
    }
}

// ---------------- generic WMMA GEMM: D = epilogue(A[M,K] * Bt[N,K]^T) ----------------
// Register-blocked: wave = 32(M) x 64(N) tile (2 A-frags reuse each B-frag -> 8 WMMA
// per K-step, 0.75KB operand traffic per WMMA). 4 waves/block -> 128 rows per block.
// gridDim.x = M/128, gridDim.y = N/64, gridDim.z = batch.
__global__ __launch_bounds__(128) void gemm_wmma(
    const h16* __restrict__ A, int lda, long long aBatch,
    const h16* __restrict__ Bt, int ldb, long long bBatch,
    int M, int N, int K,
    const float* __restrict__ S, const float* __restrict__ T, int relu,
    h16* __restrict__ D16, float* __restrict__ D32,
    long long dRow, long long dCol, long long dBatch)
{
    int lane = threadIdx.x & 31;
    int wid  = threadIdx.x >> 5;
    int m0   = (blockIdx.x * 4 + wid) * 32;
    if (m0 >= M) return;                       // wave-uniform
    int n0   = blockIdx.y * 64;
    long long z = blockIdx.z;
    const h16* Az = A + z * aBatch + (size_t)m0 * lda;
    const h16* Bz = Bt + z * bBatch;

    v8f acc[2][4] = {};
    for (int k0 = 0; k0 < K; k0 += 32) {
        if (k0 + 32 < K) {  // pull next A K-block toward the WGP (global_prefetch_b8)
            __builtin_prefetch(Az + (size_t)(lane & 15) * lda + k0 + 32, 0, 3);
            __builtin_prefetch(Az + (size_t)((lane & 15) + 16) * lda + k0 + 32, 0, 3);
        }
        v16h a0 = load_a_frag(Az, lda, k0, lane);
        v16h a1 = load_a_frag(Az + (size_t)16 * lda, lda, k0, lane);
#pragma unroll
        for (int j = 0; j < 4; ++j) {
            v16h bf = load_b_frag(Bz + (size_t)(n0 + j * 16) * ldb, ldb, k0, lane);
            acc[0][j] = __builtin_amdgcn_wmma_f32_16x16x32_f16(
                false, a0, false, bf, (short)0, acc[0][j], false, false);
            acc[1][j] = __builtin_amdgcn_wmma_f32_16x16x32_f16(
                false, a1, false, bf, (short)0, acc[1][j], false, false);
        }
    }
    int ncol = lane & 15;
    int rhi  = (lane >> 4) * 8;
#pragma unroll
    for (int i = 0; i < 2; ++i) {
#pragma unroll
        for (int j = 0; j < 4; ++j) {
#pragma unroll
            for (int r = 0; r < 8; ++r) {
                int n = n0 + j * 16 + ncol;
                int m = m0 + i * 16 + rhi + r;
                float v = acc[i][j][r] * S[n] + T[n];
                if (relu) v = fmaxf(v, 0.f);
                long long idx = z * dBatch + (long long)m * dRow + (long long)n * dCol;
                if (D32) D32[idx] = v;
                else     D16[idx] = (h16)v;
            }
        }
    }
}

extern "C" void kernel_launch(void* const* d_in, const int* in_sizes, int n_in,
                              void* d_out, int out_size, void* d_ws, size_t ws_size,
                              hipStream_t stream) {
    (void)in_sizes; (void)n_in; (void)out_size; (void)ws_size;
    const float* anchor   = (const float*)d_in[0];
    const float* positive = (const float*)d_in[1];
    const float* negative = (const float*)d_in[2];
    const float *bn0_g = (const float*)d_in[3],  *bn0_b = (const float*)d_in[4];
    const float *bn0_m = (const float*)d_in[5],  *bn0_v = (const float*)d_in[6];
    const float *conv1_w = (const float*)d_in[7], *conv1_b = (const float*)d_in[8];
    const float *bn1_g = (const float*)d_in[9],  *bn1_b = (const float*)d_in[10];
    const float *bn1_m = (const float*)d_in[11], *bn1_v = (const float*)d_in[12];
    const float *conv2_w = (const float*)d_in[13], *conv2_b = (const float*)d_in[14];
    const float *bn2_g = (const float*)d_in[15], *bn2_b = (const float*)d_in[16];
    const float *bn2_m = (const float*)d_in[17], *bn2_v = (const float*)d_in[18];
    const float *lc1_w = (const float*)d_in[19];
    const float *bn3_g = (const float*)d_in[20], *bn3_b = (const float*)d_in[21];
    const float *bn3_m = (const float*)d_in[22], *bn3_v = (const float*)d_in[23];
    const float *lc2_w = (const float*)d_in[24];
    const float *bn4_g = (const float*)d_in[25], *bn4_b = (const float*)d_in[26];
    const float *bn4_m = (const float*)d_in[27], *bn4_v = (const float*)d_in[28];
    const float *fc1_w = (const float*)d_in[29], *fc1_b = (const float*)d_in[30];
    const float *bnf1_g = (const float*)d_in[31], *bnf1_b = (const float*)d_in[32];
    const float *bnf1_m = (const float*)d_in[33], *bnf1_v = (const float*)d_in[34];
    const float *fc2_w = (const float*)d_in[35], *fc2_b = (const float*)d_in[36];
    const float *bnf2_g = (const float*)d_in[37], *bnf2_b = (const float*)d_in[38];
    const float *bnf2_m = (const float*)d_in[39], *bnf2_v = (const float*)d_in[40];
    const float *fc3_w = (const float*)d_in[41], *fc3_b = (const float*)d_in[42];
    const float *bnf3_g = (const float*)d_in[43], *bnf3_b = (const float*)d_in[44];
    const float *bnf3_m = (const float*)d_in[45], *bnf3_v = (const float*)d_in[46];
    const float *out_w = (const float*)d_in[47], *out_b = (const float*)d_in[48];

    char* ws = (char*)d_ws;
    h16* act1 = (h16*)(ws + OFF_ACT1);
    h16* act2 = (h16*)(ws + OFF_ACT2);
    h16* act3 = act1;                 // reuse: conv2 consumed act1
    h16* act4 = act2;                 // reuse: lc1 consumed act2
    h16* fcb1 = (h16*)(ws + OFF_FCB1);
    h16* fcb2 = (h16*)(ws + OFF_FCB2);
    h16* fcb3 = (h16*)(ws + OFF_FCB3);
    h16* c2w  = (h16*)(ws + OFF_C2W);
    h16* lc1w = (h16*)(ws + OFF_LC1W);
    h16* lc2w = (h16*)(ws + OFF_LC2W);
    h16* fc1w = (h16*)(ws + OFF_FC1W);
    h16* fc2w = (h16*)(ws + OFF_FC2W);
    h16* fc3w = (h16*)(ws + OFF_FC3W);
    h16* outw = (h16*)(ws + OFF_OUTW);
    float* P  = (float*)(ws + OFF_P);

    prep_params<<<1, 128, 0, stream>>>(
        bn0_g, bn0_b, bn0_m, bn0_v,
        conv1_b, bn1_g, bn1_b, bn1_m, bn1_v,
        conv2_b, bn2_g, bn2_b, bn2_m, bn2_v,
        bn3_g, bn3_b, bn3_m, bn3_v,
        bn4_g, bn4_b, bn4_m, bn4_v,
        fc1_b, bnf1_g, bnf1_b, bnf1_m, bnf1_v,
        fc2_b, bnf2_g, bnf2_b, bnf2_m, bnf2_v,
        fc3_b, bnf3_g, bnf3_b, bnf3_m, bnf3_v,
        out_b, P);

    pack_conv2<<<144, 256, 0, stream>>>(conv2_w, c2w);
    pack_lc<<<2048, 256, 0, stream>>>(lc1_w, lc1w);
    pack_lc<<<2048, 256, 0, stream>>>(lc2_w, lc2w);
    cvt_f16<<<16384, 256, 0, stream>>>(fc1_w, fc1w, 512 * 8192);
    cvt_f16<<<1024, 256, 0, stream>>>(fc2_w, fc2w, 512 * 512);
    cvt_f16<<<256, 256, 0, stream>>>(fc3_w, fc3w, 128 * 512);
    cvt_f16<<<64, 256, 0, stream>>>(out_w, outw, 128 * 128);

    // conv1 (+bn0, bn1, relu) -> act1 [(b*128+p)][c]
    conv1_kernel<<<1536, 128, 0, stream>>>(anchor, positive, negative, conv1_w, P, act1);

    // conv2 (+bn2, relu) -> act2 [p][b][c]
    conv2_wmma<<<3072, 128, 0, stream>>>(act1, c2w, P + P_S2, P + P_T2, act2);

    // lc1: 128 batched GEMMs (1536x64x64), [p][b][c] -> [p][b][c]
    dim3 gLC(12, 1, 128);
    gemm_wmma<<<gLC, 128, 0, stream>>>(act2, 64, 1536LL * 64, lc1w, 64, 4096,
        1536, 64, 64, P + P_S3, P + P_T3, 1,
        act3, nullptr, 64, 1, 1536LL * 64);

    // lc2: scatter epilogue into fc1 layout: act4[b*8192 + o*128 + p]
    gemm_wmma<<<gLC, 128, 0, stream>>>(act3, 64, 1536LL * 64, lc2w, 64, 4096,
        1536, 64, 64, P + P_S4, P + P_T4, 1,
        act4, nullptr, 8192, 128, 1);

    // fc1: 1536x512x8192
    dim3 gF1(12, 8, 1);
    gemm_wmma<<<gF1, 128, 0, stream>>>(act4, 8192, 0, fc1w, 8192, 0,
        1536, 512, 8192, P + P_SF1, P + P_TF1, 1,
        fcb1, nullptr, 512, 1, 0);

    // fc2: 1536x512x512
    gemm_wmma<<<gF1, 128, 0, stream>>>(fcb1, 512, 0, fc2w, 512, 0,
        1536, 512, 512, P + P_SF2, P + P_TF2, 1,
        fcb2, nullptr, 512, 1, 0);

    // fc3: 1536x128x512
    dim3 gF3(12, 2, 1);
    gemm_wmma<<<gF3, 128, 0, stream>>>(fcb2, 512, 0, fc3w, 512, 0,
        1536, 128, 512, P + P_SF3, P + P_TF3, 1,
        fcb3, nullptr, 128, 1, 0);

    // out: 1536x128x128 -> d_out (f32); rows are [anchor|positive|negative] concatenated
    gemm_wmma<<<gF3, 128, 0, stream>>>(fcb3, 128, 0, outw, 128, 0,
        1536, 128, 128, P + P_SO, P + P_TO, 0,
        nullptr, (float*)d_out, 128, 1, 0);
}
